// KAN_Convolutional_Layer_76922864271830
// MI455X (gfx1250) — compile-verified, matrix operand order checked by reading, not run on hardware
//
#include <hip/hip_runtime.h>
#include <hip/hip_bf16.h>

typedef __attribute__((ext_vector_type(16))) _Float16 v16h;
typedef __attribute__((ext_vector_type(8)))  float    v8f;

#define CIN    32
#define COUT   64
#define KPOS   9          // 3x3 kernel positions
#define NB     8          // spline bases per value
#define FPP    10         // features per (cin,k) pair: silu + 8 bases + 1 zero pad
#define NPAIR  (CIN*KPOS) // 288
#define KPAD   (NPAIR*FPP)// 2880 (divisible by 32)
#define NCHUNK (KPAD/32)  // 90 K-chunks of 32
#define NTILES (COUT/16)  // 4 N-tiles
#define ROWS   64         // output pixels per workgroup
#define LTOT   4096       // 64*64 pixels per image
#define HW     64

__device__ __forceinline__ unsigned pk2(float a, float b) {
  union { _Float16 h[2]; unsigned u; } u_;
  u_.h[0] = (_Float16)a; u_.h[1] = (_Float16)b;
  return u_.u;
}

// ---------------------------------------------------------------------------
// Pack base_w and spline_w*scaler into one f16 weight matrix, stored directly
// in WMMA B-fragment order: fragment f = (chunk*NTILES + ntile), 32 lanes x
// 16 halfs (32B) per lane.  B element (k_local, n): lanes 0-15 hold K 0-7,16-23,
// lanes 16-31 hold K 8-15,24-31; half h<8 -> K = grp*8+h, h>=8 -> K = 16+grp*8+h-8.
// ---------------------------------------------------------------------------
__global__ __launch_bounds__(256) void kan_pack_weights(
    const float* __restrict__ base_w, const float* __restrict__ spline_w,
    const float* __restrict__ scaler, _Float16* __restrict__ wpack) {
  int t = blockIdx.x * 256 + threadIdx.x;
  if (t >= NCHUNK * NTILES * 32) return;
  int lane  = t & 31;
  int nt    = (t >> 5) & (NTILES - 1);
  int chunk = t >> 7;
  int nloc = lane & 15, hiL = lane >> 4;
  int co = nt * 16 + nloc;
  v16h v;
#pragma unroll
  for (int hh = 0; hh < 16; ++hh) {
    int kIn32 = (hh < 8) ? (hiL * 8 + hh) : (16 + hiL * 8 + (hh - 8));
    int kp   = chunk * 32 + kIn32;          // padded feature index, < 2880
    int pair = kp / FPP;
    int j    = kp - pair * FPP;             // 0=silu, 1..8=spline, 9=pad
    int cin  = pair / KPOS, kk = pair - cin * KPOS;
    int idx  = (co * CIN + cin) * KPOS + kk;
    float w = 0.0f;
    if (j == 0)      w = base_w[idx];
    else if (j <= NB) w = spline_w[idx * NB + (j - 1)] * scaler[idx];
    v[hh] = (_Float16)w;
  }
  *(v16h*)(wpack + (size_t)t * 16) = v;
}

// ---------------------------------------------------------------------------
// Fused: patch extraction + silu + B-spline basis (uniform grid, Cox-de Boor
// fully unrolled with constant knots) into LDS f16 features, then WMMA GEMM
// against the packed weights.  Each workgroup: 64 rows x 64 cols of output.
// Wave w: M-tile = w>>1 (16 rows), N-tiles = {2*(w&1), 2*(w&1)+1}.
// ---------------------------------------------------------------------------
__global__ __launch_bounds__(256) void kan_conv_gemm(
    const float* __restrict__ x, const _Float16* __restrict__ wpack,
    float* __restrict__ out) {
  __shared__ _Float16 Fslab[ROWS * (32 * FPP)];   // 64 * 320 halfs = 40 KB

  const int tid  = threadIdx.x;
  const int lane = tid & 31, wave = tid >> 5;
  const int rowBaseG = blockIdx.x * ROWS;         // global pixel row
  const int b  = rowBaseG >> 12;                  // image index (L = 4096)
  const int l0 = rowBaseG & (LTOT - 1);

  const int mtile  = wave >> 1;                   // 0..3
  const int ntBase = (wave & 1) * 2;              // 0 or 2
  const int m  = lane & 15, hi = lane >> 4;

  v8f acc0 = {};
  v8f acc1 = {};

  const float h = 0.4f;                           // (1-(-1))/grid_size

  for (int s = 0; s < 9; ++s) {                   // 9 slabs of 32 pairs
    // ---- cooperative feature fill: 64 rows x 32 (cin,k) pairs ----
#pragma unroll
    for (int i = 0; i < 8; ++i) {
      int e   = tid + 256 * i;                    // 2048 tasks
      int row = e >> 5, pl = e & 31;
      int pair = s * 32 + pl;
      int cin = pair / KPOS, kk = pair - cin * KPOS;
      int l  = l0 + row;
      int yy = l >> 6, xx = l & 63;
      int iy = yy + (kk / 3) - 1;
      int ix = xx + (kk % 3) - 1;
      float p = 0.0f;
      if ((unsigned)iy < (unsigned)HW && (unsigned)ix < (unsigned)HW)
        p = x[(((size_t)(b * CIN + cin) * HW + iy) << 6) + ix];
      // silu
      float sil = p * (1.0f / (1.0f + __expf(-p)));
      // B-spline bases, uniform knots t_q = -1 + (q-3)*h, q = 0..11
      float t_[12];
#pragma unroll
      for (int q = 0; q < 12; ++q) t_[q] = -1.0f + (float)(q - 3) * h;
      float bas[11];
#pragma unroll
      for (int q = 0; q < 11; ++q)
        bas[q] = (p >= t_[q] && p < t_[q + 1]) ? 1.0f : 0.0f;
#pragma unroll
      for (int j = 1; j <= 3; ++j) {
        float inv = 1.0f / ((float)j * h);
#pragma unroll
        for (int q = 0; q < 11 - j; ++q)
          bas[q] = (p - t_[q]) * inv * bas[q] +
                   (t_[q + j + 1] - p) * inv * bas[q + 1];
      }
      // 10 f16 features, packed as 5 aligned dword stores
      unsigned* dst = (unsigned*)&Fslab[row * (32 * FPP) + pl * FPP];
      dst[0] = pk2(sil,    bas[0]);
      dst[1] = pk2(bas[1], bas[2]);
      dst[2] = pk2(bas[3], bas[4]);
      dst[3] = pk2(bas[5], bas[6]);
      dst[4] = pk2(bas[7], 0.0f);
    }
    __syncthreads();

    // ---- WMMA over this slab's 10 K-chunks ----
    const _Float16* rowp = &Fslab[(mtile * 16 + m) * (32 * FPP)];
#pragma unroll
    for (int c = 0; c < 10; ++c) {
      int fo = c * 32 + hi * 8;                   // A-fragment K offsets
      v16h a;
      ((uint4*)&a)[0] = *(const uint4*)(rowp + fo);        // K 0-7 / 8-15
      ((uint4*)&a)[1] = *(const uint4*)(rowp + fo + 16);   // K 16-23 / 24-31
      int chunkG = s * 10 + c;
      const v16h* wb = (const v16h*)wpack;
      v16h b0 = wb[(size_t)(chunkG * NTILES + ntBase)     * 32 + lane];
      v16h b1 = wb[(size_t)(chunkG * NTILES + ntBase + 1) * 32 + lane];
      acc0 = __builtin_amdgcn_wmma_f32_16x16x32_f16(
                 false, a, false, b0, (short)0, acc0, false, false);
      acc1 = __builtin_amdgcn_wmma_f32_16x16x32_f16(
                 false, a, false, b1, (short)0, acc1, false, false);
    }
    __syncthreads();
  }

  // ---- store: C/D layout -> out[b][co][l] ----
#pragma unroll
  for (int v = 0; v < 8; ++v) {
    int mrow = mtile * 16 + hi * 8 + v;
    int l    = l0 + mrow;
    int co0  = ntBase * 16 + m;
    int co1  = (ntBase + 1) * 16 + m;
    out[((size_t)(b * COUT + co0) << 12) + l] = acc0[v];
    out[((size_t)(b * COUT + co1) << 12) + l] = acc1[v];
  }
}

extern "C" void kernel_launch(void* const* d_in, const int* in_sizes, int n_in,
                              void* d_out, int out_size, void* d_ws, size_t ws_size,
                              hipStream_t stream) {
  (void)in_sizes; (void)n_in; (void)out_size; (void)ws_size;
  const float* x        = (const float*)d_in[0];
  const float* base_w   = (const float*)d_in[1];
  const float* spline_w = (const float*)d_in[2];
  const float* scaler   = (const float*)d_in[3];
  float* out = (float*)d_out;
  _Float16* wpack = (_Float16*)d_ws;              // needs 90*4*32*32 B = 368,640 B

  int packThreads = NCHUNK * NTILES * 32;         // 11520
  kan_pack_weights<<<(packThreads + 255) / 256, 256, 0, stream>>>(
      base_w, spline_w, scaler, wpack);

  int nBlocks = (16 * LTOT) / ROWS;               // 1024 workgroups
  kan_conv_gemm<<<nBlocks, 256, 0, stream>>>(x, wpack, out);
}
